// SimAttention_68272800137484
// MI455X (gfx1250) — matrix operand order repaired and checked by reading, not compiled, not run on hardware
//
#include <hip/hip_runtime.h>

typedef _Float16 f16;
typedef _Float16 v16h __attribute__((ext_vector_type(16)));
typedef float    v8f  __attribute__((ext_vector_type(8)));

#define N_ 2
#define L_ 2048
#define S_ 2048
#define H_ 8
#define D_ 64
#define STILE 64
#define LROWS 128          // L rows per workgroup (8 waves x 16)
#define KPAD  88           // padded LDS row stride in f16 (176 B)

// ---------------- Phase 1: feature map  Phi(x)=elu(x)+1, masked, f32 -> f16,
// relaid out [n][h][l][d] so each GEMM row is 128 contiguous bytes. ----------
__global__ __launch_bounds__(256)
void featmap_kernel(const float* __restrict__ q, const float* __restrict__ k,
                    const float* __restrict__ qm, const float* __restrict__ km,
                    f16* __restrict__ qf, f16* __restrict__ kf)
{
    int i = blockIdx.x * 256 + threadIdx.x;        // 0 .. N*L*H*D-1
    int d = i & (D_ - 1);
    int h = (i >> 6) & (H_ - 1);
    int l = (i >> 9) & (L_ - 1);
    int n = i >> 20;
    int o = (((n * H_ + h) * L_ + l) << 6) + d;    // [n][h][l][d]
    float mq = qm[n * L_ + l];
    float mk = km[n * S_ + l];
    float xq = q[i], xk = k[i];
    float fq = (xq > 0.f ? xq + 1.f : expf(xq)) * mq;   // elu(x)+1
    float fk = (xk > 0.f ? xk + 1.f : expf(xk)) * mk;
    qf[o] = (f16)fq;
    kf[o] = (f16)fk;
}

// ---------------- Phase 2: fused streaming GEMM + row max/mean + gate -------
// Double-buffered LDS staging: one barrier per 64-row K tile; global loads for
// tile i+1 are issued right after the barrier and waited on a full tile later.
__global__ __launch_bounds__(256)
void qkstat_kernel(const f16* __restrict__ qf, const f16* __restrict__ kf,
                   const float* __restrict__ queries, const float* __restrict__ qm,
                   const float* __restrict__ W, const float* __restrict__ bias,
                   float* __restrict__ out)
{
    __shared__ f16   kt[2 * STILE * KPAD];   // 2 x (64 K-rows, padded stride)
    __shared__ float gate[LROWS];

    const int blk  = blockIdx.x;
    const int tile = blk & 15;           // L/128 = 16 tiles
    const int h    = (blk >> 4) & 7;
    const int n    = blk >> 7;
    const int tid  = threadIdx.x;
    const int lane = tid & 31;
    const int wave = tid >> 5;
    const int half = lane >> 4;
    const int lrow = lane & 15;

    const f16* qbase = qf + ((size_t)(n * H_ + h)) * L_ * D_;
    const f16* kbase = kf + ((size_t)(n * H_ + h)) * S_ * D_;

    // ---- A registers (loop-invariant): 16x64 Phi(Q) strip per wave ----
    // ISA 16-bit A layout: v[0..7] <- K = half*8 + 0..7,  v[8..15] <- K = 16+half*8+0..7
    const int row0 = tile * LROWS + wave * 16;
    const uint4* q4 = (const uint4*)(qbase + (size_t)(row0 + lrow) * D_); // 8 x 16B per row
    union { v16h v; uint4 u[2]; } a0u, a1u;
    a0u.u[0] = q4[half];     a0u.u[1] = q4[2 + half];   // K 0..31
    a1u.u[0] = q4[4 + half]; a1u.u[1] = q4[6 + half];   // K 32..63
    const v16h a0 = a0u.v, a1 = a1u.v;

    // this thread's 4 staging slots: 64 rows x 16 u64 segments / 256 threads
    const int srow[4] = { tid >> 4, (tid + 256) >> 4, (tid + 512) >> 4, (tid + 768) >> 4 };
    const int sseg    = tid & 15;

    // prologue: fetch tile 0 into registers
    unsigned long long rstage[4];
#pragma unroll
    for (int j = 0; j < 4; ++j)
        rstage[j] = *((const unsigned long long*)(kbase + (size_t)srow[j] * D_) + sseg);

    float mx[8], sm[8];
#pragma unroll
    for (int r = 0; r < 8; ++r) { mx[r] = -3.4e38f; sm[r] = 0.f; }

    int cur = 0;
    for (int s0 = 0; s0 < S_; s0 += STILE) {
        f16* buf = kt + cur * (STILE * KPAD);
        // publish staged tile to LDS
#pragma unroll
        for (int j = 0; j < 4; ++j)
            ((unsigned long long*)buf)[srow[j] * (KPAD / 4) + sseg] = rstage[j];
        __syncthreads();

        // issue global loads for the NEXT tile; latency hidden behind the WMMAs below
        if (s0 + STILE < S_) {
#pragma unroll
            for (int j = 0; j < 4; ++j)
                rstage[j] = *((const unsigned long long*)
                              (kbase + (size_t)(s0 + STILE + srow[j]) * D_) + sseg);
        }

#pragma unroll
        for (int ss = 0; ss < STILE / 16; ++ss) {
            // ISA 16-bit B layout: lane = column (S row), v[0..7] <- K = half*8 + 0..7, v[8..15] <- K = 16+half*8+0..7
            const uint4* k4 = (const uint4*)(buf + (ss * 16 + lrow) * KPAD);
            union { v16h v; uint4 u[2]; } b0u, b1u;
            b0u.u[0] = k4[half];     b0u.u[1] = k4[2 + half]; // K 0..31
            b1u.u[0] = k4[4 + half]; b1u.u[1] = k4[6 + half]; // K 32..63
            v8f c = {};
            c = __builtin_amdgcn_wmma_f32_16x16x32_f16(false, a0, false, b0u.v,
                                                       (short)0, c, false, false);
            c = __builtin_amdgcn_wmma_f32_16x16x32_f16(false, a1, false, b1u.v,
                                                       (short)0, c, false, false);
#pragma unroll
            for (int r = 0; r < 8; ++r) { mx[r] = fmaxf(mx[r], c[r]); sm[r] += c[r]; }
        }
        cur ^= 1;
    }

    // ---- reduce across the 16 lanes holding each row (C layout: row M in VGPR r) ----
#pragma unroll
    for (int m = 1; m < 16; m <<= 1) {
#pragma unroll
        for (int r = 0; r < 8; ++r) {
            mx[r] = fmaxf(mx[r], __shfl_xor(mx[r], m, 32));
            sm[r] += __shfl_xor(sm[r], m, 32);
        }
    }

    if (lrow == 0) {
        float w0 = W[0], w1 = W[1], bb = bias[0];
#pragma unroll
        for (int r = 0; r < 8; ++r) {
            int row = wave * 16 + half * 8 + r;          // lanes 0/16 cover rows 0-7 / 8-15
            float mean = sm[r] * (1.f / (float)S_);
            float g = w0 * mean + w1 * mx[r] + bb;
            gate[row] = 1.f / (1.f + expf(-g));
        }
    }
    __syncthreads();

    // ---- output: queries * q_mask * gate, coalesced float4 ----
    for (int i = tid; i < LROWS * (D_ / 4); i += 256) {
        int row = i >> 4, c4 = i & 15;
        int l = tile * LROWS + row;
        size_t gidx = ((((size_t)n * L_ + l) * H_ + h) << 6) + c4 * 4;
        float4 v = *(const float4*)(queries + gidx);
        float s = qm[n * L_ + l] * gate[row];
        v.x *= s; v.y *= s; v.z *= s; v.w *= s;
        *(float4*)(out + gidx) = v;
    }
}

extern "C" void kernel_launch(void* const* d_in, const int* in_sizes, int n_in,
                              void* d_out, int out_size, void* d_ws, size_t ws_size,
                              hipStream_t stream) {
    const float* q  = (const float*)d_in[0];
    const float* k  = (const float*)d_in[1];
    // d_in[2] (values) is unused by the reference's output
    const float* qm = (const float*)d_in[3];
    const float* km = (const float*)d_in[4];
    const float* W  = (const float*)d_in[5];
    const float* b  = (const float*)d_in[6];
    float* out = (float*)d_out;

    f16* qf = (f16*)d_ws;                               // 4 MB
    f16* kf = qf + (size_t)N_ * H_ * L_ * D_;           // 4 MB

    const int total = N_ * L_ * H_ * D_;                // 2,097,152
    featmap_kernel<<<total / 256, 256, 0, stream>>>(q, k, qm, km, qf, kf);
    qkstat_kernel<<<N_ * H_ * (L_ / LROWS), 256, 0, stream>>>(qf, kf, q, qm, W, b, out);
}
